// SelfAttention_33767032881830
// MI455X (gfx1250) — matrix operand order repaired and checked by reading, not compiled, not run on hardware
//
#include <hip/hip_runtime.h>

#define N_SEQ 8192
#define DIM   256

typedef __attribute__((ext_vector_type(16))) __bf16 v16bf;
typedef __attribute__((ext_vector_type(8)))  __bf16 v8bf;
typedef __attribute__((ext_vector_type(8)))  float  v8f;

#define WMMA_BF16(A, B, C) \
    __builtin_amdgcn_wmma_f32_16x16x32_bf16(false, (A), false, (B), (short)0, (C), false, false)

static __device__ inline v8f zero_v8f() {
    v8f z = {0.f, 0.f, 0.f, 0.f, 0.f, 0.f, 0.f, 0.f};
    return z;
}

// ---------------------------------------------------------------------------
// Kernel 0: pack W (f32 [D][D], [in][out]) into per-lane WMMA B-fragment
// layout, bf16.  Layout: [w][ntile t][kstep s][lane][e] ; value =
// W[(32s + e + (lane>=16?16:0))][16t + lane%16].
// One thread per (w,t,s,lane), each writes 16 contiguous bf16 (32 B).
// ---------------------------------------------------------------------------
__global__ __launch_bounds__(256) void pack_w(const float* __restrict__ Wq,
                                              const float* __restrict__ Wk,
                                              const float* __restrict__ Wv,
                                              __bf16* __restrict__ wpack) {
    int id = blockIdx.x * 256 + threadIdx.x;     // 0 .. 12287
    int lane = id & 31;
    int s    = (id >> 5) & 7;
    int t    = (id >> 8) & 15;
    int w    = id >> 12;                         // 0..2
    const float* W = (w == 0) ? Wq : ((w == 1) ? Wk : Wv);
    int col  = t * 16 + (lane & 15);
    int krow = s * 32 + ((lane >= 16) ? 16 : 0);
    __bf16* dst = wpack + (size_t)id * 16;
#pragma unroll
    for (int e = 0; e < 16; ++e)
        dst[e] = (__bf16)W[(size_t)(krow + e) * DIM + col];
}

// ---------------------------------------------------------------------------
// Kernel 1: QKV projection.  Each block: one 16-row m-tile of one projection.
// 8 waves x 2 n-tiles = 16 n-tiles (all 256 output columns).
// Q, K stored row-major bf16; V stored transposed (Vt[d][j]) bf16.
// ---------------------------------------------------------------------------
__global__ __launch_bounds__(256) void qkv_proj(const float* __restrict__ x,
                                                const __bf16* __restrict__ wpack,
                                                __bf16* __restrict__ Qb,
                                                __bf16* __restrict__ Kb,
                                                __bf16* __restrict__ Vt) {
    int w   = blockIdx.x % 3;
    int mt  = blockIdx.x / 3;            // 0..511
    int wid  = threadIdx.x >> 5;
    int lane = threadIdx.x & 31;
    int hi = lane >> 4, ln = lane & 15;
    int i0 = mt * 16;

    // Preload A fragments for the 16 rows (convert f32 -> bf16 in-register).
    v16bf a[8];
    const float* xrow = x + (size_t)(i0 + ln) * DIM;
#pragma unroll
    for (int ks = 0; ks < 8; ++ks) {
        int k0 = ks * 32 + hi * 16;
        v16bf av;
#pragma unroll
        for (int e = 0; e < 8; ++e) av[e]     = (__bf16)xrow[k0 + e];
#pragma unroll
        for (int e = 0; e < 8; ++e) av[8 + e] = (__bf16)xrow[k0 + 8 + e];
        a[ks] = av;
    }

    const __bf16* wp = wpack + (size_t)w * 65536;
#pragma unroll
    for (int ti = 0; ti < 2; ++ti) {
        int t = wid * 2 + ti;
        v8f acc = zero_v8f();
#pragma unroll
        for (int ks = 0; ks < 8; ++ks) {
            v16bf b = *(const v16bf*)(wp + ((size_t)(t * 8 + ks) * 32 + lane) * 16);
            acc = WMMA_BF16(a[ks], b, acc);
        }
        if (w < 2) {
            __bf16* out = (w == 0) ? Qb : Kb;
            int col = t * 16 + ln;
#pragma unroll
            for (int r = 0; r < 8; ++r)
                out[(size_t)(i0 + r + hi * 8) * DIM + col] = (__bf16)acc[r];
        } else {
            v8bf pv;
#pragma unroll
            for (int r = 0; r < 8; ++r) pv[r] = (__bf16)acc[r];
            *(v8bf*)(Vt + (size_t)(t * 16 + ln) * N_SEQ + i0 + hi * 8) = pv;
        }
    }
}

// ---------------------------------------------------------------------------
// Kernel 2: flash attention.  Block = 64 query rows, 8 waves.
// Wave w: m-tile m = w&3 (16 rows), key chunk s = w>>2 (32 of 64 cols).
// Online softmax stats in LDS; P transposed via 8 KB LDS buffer.
// ---------------------------------------------------------------------------
__global__ __launch_bounds__(256) void flash_attn(const __bf16* __restrict__ Qb,
                                                  const __bf16* __restrict__ Kb,
                                                  const __bf16* __restrict__ Vt,
                                                  float* __restrict__ out) {
    __shared__ __align__(32) unsigned char smem_raw[64 * 64 * 2]; // P (bf16) / combine (f32)
    __shared__ float lds_m[64], lds_l[64];
    __shared__ float lds_pmax[2][64], lds_psum[2][64];
    __bf16 (*Pb)[64] = (__bf16 (*)[64])smem_raw;
    float* Cb = (float*)smem_raw;

    int tid  = threadIdx.x;
    int wid  = tid >> 5;
    int lane = tid & 31;
    int hi = lane >> 4, ln = lane & 15;
    int m = wid & 3, s = wid >> 2;
    int i0   = blockIdx.x * 64;
    int mrow = i0 + m * 16;

    // Preload Q A-fragments for this wave's 16 rows (whole D = 8 k-steps).
    v16bf aq[8];
    const __bf16* qrow = Qb + (size_t)(mrow + ln) * DIM;
#pragma unroll
    for (int ks = 0; ks < 8; ++ks) {
        int k0 = ks * 32 + hi * 16;
        v8bf c0 = *(const v8bf*)(qrow + k0);
        v8bf c1 = *(const v8bf*)(qrow + k0 + 8);
        v16bf av;
#pragma unroll
        for (int e = 0; e < 8; ++e) { av[e] = c0[e]; av[8 + e] = c1[e]; }
        aq[ks] = av;
    }

    v8f O[16];
#pragma unroll
    for (int dt = 0; dt < 16; ++dt) O[dt] = zero_v8f();

    if (tid < 64) { lds_m[tid] = -1e30f; lds_l[tid] = 0.f; }
    __syncthreads();

    const float scale = 0.0625f; // 1/sqrt(256)

    for (int j0 = 0; j0 < N_SEQ; j0 += 64) {
        // ---- S tiles: this wave's 2 n-tiles (cols j0 + s*32 .. +31) -------
        v8f S0 = zero_v8f(), S1 = zero_v8f();
        {
            const __bf16* kr0 = Kb + (size_t)(j0 + s * 32 + ln) * DIM + hi * 16;
            const __bf16* kr1 = Kb + (size_t)(j0 + s * 32 + 16 + ln) * DIM + hi * 16;
#pragma unroll
            for (int ks = 0; ks < 8; ++ks) {
                v16bf b0 = *(const v16bf*)(kr0 + ks * 32);
                S0 = WMMA_BF16(aq[ks], b0, S0);
            }
#pragma unroll
            for (int ks = 0; ks < 8; ++ks) {
                v16bf b1 = *(const v16bf*)(kr1 + ks * 32);
                S1 = WMMA_BF16(aq[ks], b1, S1);
            }
        }
#pragma unroll
        for (int r = 0; r < 8; ++r) { S0[r] *= scale; S1[r] *= scale; }

        // ---- row max over this wave's 32 cols -----------------------------
        float rmax[8];
#pragma unroll
        for (int r = 0; r < 8; ++r) rmax[r] = fmaxf(S0[r], S1[r]);
#pragma unroll
        for (int mask = 1; mask < 16; mask <<= 1)
#pragma unroll
            for (int r = 0; r < 8; ++r)
                rmax[r] = fmaxf(rmax[r], __shfl_xor(rmax[r], mask, 32));
        if (ln == 0)
#pragma unroll
            for (int r = 0; r < 8; ++r)
                lds_pmax[s][m * 16 + r + hi * 8] = rmax[r];
        __syncthreads(); // A

        // ---- new max, rescale factor, P = exp(S - mnew), row sums ---------
        float alpha[8], mnew[8];
#pragma unroll
        for (int r = 0; r < 8; ++r) {
            int row = m * 16 + r + hi * 8;
            float mo = lds_m[row];
            float mn = fmaxf(mo, fmaxf(lds_pmax[0][row], lds_pmax[1][row]));
            mnew[r] = mn;
            alpha[r] = __expf(mo - mn);
        }
        float rs[8];
#pragma unroll
        for (int r = 0; r < 8; ++r) {
            S0[r] = __expf(S0[r] - mnew[r]);
            S1[r] = __expf(S1[r] - mnew[r]);
            rs[r] = S0[r] + S1[r];
        }
#pragma unroll
        for (int mask = 1; mask < 16; mask <<= 1)
#pragma unroll
            for (int r = 0; r < 8; ++r)
                rs[r] += __shfl_xor(rs[r], mask, 32);
        if (ln == 0)
#pragma unroll
            for (int r = 0; r < 8; ++r)
                lds_psum[s][m * 16 + r + hi * 8] = rs[r];

        // ---- write P (bf16) into LDS for layout transpose -----------------
#pragma unroll
        for (int r = 0; r < 8; ++r) {
            int row = m * 16 + r + hi * 8;
            Pb[row][s * 32 + ln]      = (__bf16)S0[r];
            Pb[row][s * 32 + 16 + ln] = (__bf16)S1[r];
        }

        // ---- rescale running output ---------------------------------------
#pragma unroll
        for (int dt = 0; dt < 16; ++dt)
#pragma unroll
            for (int r = 0; r < 8; ++r) O[dt][r] *= alpha[r];
        __syncthreads(); // B

        // ---- update global row stats (rows 0..63 by first two waves) ------
        if (tid < 64) {
            float mo = lds_m[tid];
            float mn = fmaxf(mo, fmaxf(lds_pmax[0][tid], lds_pmax[1][tid]));
            lds_l[tid] = __expf(mo - mn) * lds_l[tid] +
                         lds_psum[0][tid] + lds_psum[1][tid];
            lds_m[tid] = mn;
        }

        // ---- O += P(m, k-chunk s) @ V(k-chunk s, :) -----------------------
        v16bf ap;
        {
            const __bf16* prow = &Pb[m * 16 + ln][s * 32 + hi * 16];
            v8bf p0 = *(const v8bf*)(prow);
            v8bf p1 = *(const v8bf*)(prow + 8);
#pragma unroll
            for (int e = 0; e < 8; ++e) { ap[e] = p0[e]; ap[8 + e] = p1[e]; }
        }
        const __bf16* vbase = Vt + (size_t)ln * N_SEQ + j0 + s * 32 + hi * 16;
#pragma unroll
        for (int dt = 0; dt < 16; ++dt) {
            v16bf bv = *(const v16bf*)(vbase + (size_t)dt * 16 * N_SEQ);
            O[dt] = WMMA_BF16(ap, bv, O[dt]);
        }
        __syncthreads(); // C
    }

    // ---- final: combine wave pairs (s=0,1), divide by l, store ------------
    float lrow[8];
#pragma unroll
    for (int r = 0; r < 8; ++r) lrow[r] = lds_l[m * 16 + r + hi * 8];

    for (int chunk = 0; chunk < 8; ++chunk) {
        if (s == 1) {
#pragma unroll
            for (int dtl = 0; dtl < 2; ++dtl) {
                int dt = chunk * 2 + dtl;
#pragma unroll
                for (int r = 0; r < 8; ++r)
                    Cb[(m * 2 + dtl) * 256 + (r + hi * 8) * 16 + ln] = O[dt][r];
            }
        }
        __syncthreads();
        if (s == 0) {
#pragma unroll
            for (int dtl = 0; dtl < 2; ++dtl) {
                int dt = chunk * 2 + dtl;
#pragma unroll
                for (int r = 0; r < 8; ++r) {
                    float v = O[dt][r] +
                              Cb[(m * 2 + dtl) * 256 + (r + hi * 8) * 16 + ln];
                    out[(size_t)(mrow + r + hi * 8) * DIM + dt * 16 + ln] =
                        v / lrow[r];
                }
            }
        }
        __syncthreads();
    }
}

// ---------------------------------------------------------------------------
extern "C" void kernel_launch(void* const* d_in, const int* in_sizes, int n_in,
                              void* d_out, int out_size, void* d_ws, size_t ws_size,
                              hipStream_t stream) {
    const float* x  = (const float*)d_in[0];
    const float* Wq = (const float*)d_in[1];
    const float* Wk = (const float*)d_in[2];
    const float* Wv = (const float*)d_in[3];
    float* out = (float*)d_out;

    char* ws = (char*)d_ws;
    const size_t WPACK_BYTES = 3ull * 16 * 8 * 32 * 16 * 2; // 393216
    const size_t MAT_BYTES   = (size_t)N_SEQ * DIM * 2;     // 4 MB each
    __bf16* wpack = (__bf16*)(ws);
    __bf16* Qb    = (__bf16*)(ws + WPACK_BYTES);
    __bf16* Kb    = (__bf16*)(ws + WPACK_BYTES + MAT_BYTES);
    __bf16* Vt    = (__bf16*)(ws + WPACK_BYTES + 2 * MAT_BYTES);

    pack_w<<<48, 256, 0, stream>>>(Wq, Wk, Wv, wpack);
    qkv_proj<<<(N_SEQ / 16) * 3, 256, 0, stream>>>(x, wpack, Qb, Kb, Vt);
    flash_attn<<<N_SEQ / 64, 256, 0, stream>>>(Qb, Kb, Vt, out);
}
